// MultiQueryAttention_42039139893548
// MI455X (gfx1250) — matrix operand order repaired and checked by reading, not compiled, not run on hardware
//
#include <hip/hip_runtime.h>
#include <hip/hip_bf16.h>

typedef __attribute__((ext_vector_type(16))) _Float16 v16h;
typedef __attribute__((ext_vector_type(8)))  _Float16 v8h;
typedef __attribute__((ext_vector_type(8)))  float    v8f;

constexpr int Bn  = 4;
constexpr int Sn  = 1024;
constexpr int Dn  = 1024;
constexpr int Hn  = 16;
constexpr int DKn = 64;

// ---------- workspace layout (bytes) ----------
constexpr size_t OFF_WQT = 0;                                    // [H][DK][D] f16
constexpr size_t OFF_WKT = OFF_WQT + (size_t)Hn*DKn*Dn*2;        // [DK][D]    f16
constexpr size_t OFF_WVT = OFF_WKT + (size_t)DKn*Dn*2;           // [DK][D]    f16
constexpr size_t OFF_WCT = OFF_WVT + (size_t)DKn*Dn*2;           // [DK][H*DK] f16
constexpr size_t OFF_QH  = OFF_WCT + (size_t)DKn*Hn*DKn*2;       // [B][H][S][DK] f16
constexpr size_t OFF_K   = OFF_QH  + (size_t)Bn*Hn*Sn*DKn*2;     // [B][S][DK] f16
constexpr size_t OFF_VT  = OFF_K   + (size_t)Bn*Sn*DKn*2;        // [B][DK][S] f16
constexpr size_t OFF_CC  = OFF_VT  + (size_t)Bn*DKn*Sn*2;        // [B][S][H*DK] f16
// total ~19.4 MB

// ---------- WMMA helpers ----------
__device__ inline v8f wmma_f32_f16(v16h a, v16h b, v8f c) {
  // D = A(16x32 f16) * B(32x16 f16) + C(16x16 f32)
  return __builtin_amdgcn_wmma_f32_16x16x32_f16(false, a, false, b, (short)0, c,
                                                false, false);
}

__device__ inline v8f zero8() {
  v8f z;
#pragma unroll
  for (int i = 0; i < 8; ++i) z[i] = 0.0f;
  return z;
}

// A fragment, 16-bit, 16x32: lane(row=lane&15) chunk0 @ K=hi8..hi8+7, chunk1 @ K=hi8+16..hi8+23
__device__ inline v16h load_a_f16(const _Float16* row, int k0, int hi8) {
  v8h lo = *(const v8h*)(row + k0 + hi8);
  v8h hi = *(const v8h*)(row + k0 + hi8 + 16);
  return __builtin_shufflevector(lo, hi, 0,1,2,3,4,5,6,7,8,9,10,11,12,13,14,15);
}

// Same A fragment but source is fp32 row-major (convert in registers)
__device__ inline v16h load_a_f32(const float* row, int k0, int hi8) {
  const float* p0 = row + k0 + hi8;
  const float* p1 = p0 + 16;
  v16h a;
#pragma unroll
  for (int i = 0; i < 8; ++i) { a[i] = (_Float16)p0[i]; a[i+8] = (_Float16)p1[i]; }
  return a;
}

// B fragment, 16-bit, 32x16 from B^T row-major (row = output column):
// lane half h holds K=8h..8h+7 and K=16+8h..16+8h+7 (matches A) -> two 16B loads
__device__ inline v16h load_b_f16(const _Float16* colrow, int k0, int koffB) {
  v8h lo = *(const v8h*)(colrow + k0 + (koffB >> 1));
  v8h hi = *(const v8h*)(colrow + k0 + (koffB >> 1) + 16);
  return __builtin_shufflevector(lo, hi, 0,1,2,3,4,5,6,7,8,9,10,11,12,13,14,15);
}

// ---------- prep: transpose fp32 [batch][R][C] -> f16 [batch][C][R] ----------
__global__ void transpose_to_f16_kernel(const float* __restrict__ src,
                                        _Float16* __restrict__ dst,
                                        int R, int C, long long total) {
  long long o = (long long)blockIdx.x * blockDim.x + threadIdx.x;
  if (o >= total) return;
  long long rc  = (long long)R * C;
  long long bb  = o / rc;
  long long rem = o - bb * rc;
  int r = (int)(rem % R);
  int c = (int)(rem / R);
  dst[o] = (_Float16)src[bb * rc + (long long)r * C + c];
}

// ---------- Q projection: qh[b,h,s,:] = query[b,s,:] @ Wq[h] + bq[h] ----------
__global__ __launch_bounds__(128) void qproj_kernel(const float* __restrict__ query,
                                                    const _Float16* __restrict__ WqT,
                                                    const float* __restrict__ bq,
                                                    _Float16* __restrict__ qh) {
  int lane = threadIdx.x & 31;
  int w    = blockIdx.x * 4 + (threadIdx.x >> 5);   // 0..4095
  int st   = w & 63;
  int h    = (w >> 6) & 15;
  int b    = w >> 10;
  int l15  = lane & 15;
  int hi8  = (lane & 16) ? 8 : 0;
  int kb   = (lane & 16) ? 16 : 0;

  const float* Arow = query + ((size_t)b * Sn + st * 16 + l15) * Dn;
  const _Float16* Bcol[4];
#pragma unroll
  for (int nt = 0; nt < 4; ++nt)
    Bcol[nt] = WqT + ((size_t)h * DKn + nt * 16 + l15) * Dn;

  v8f acc[4];
#pragma unroll
  for (int nt = 0; nt < 4; ++nt) acc[nt] = zero8();

  for (int k0 = 0; k0 < Dn; k0 += 32) {
    v16h a = load_a_f32(Arow, k0, hi8);
#pragma unroll
    for (int nt = 0; nt < 4; ++nt)
      acc[nt] = wmma_f32_f16(a, load_b_f16(Bcol[nt], k0, kb), acc[nt]);
  }

#pragma unroll
  for (int nt = 0; nt < 4; ++nt) {
    float bias = bq[h * DKn + nt * 16 + l15];
#pragma unroll
    for (int r = 0; r < 8; ++r) {
      int m = r + hi8;
      qh[((size_t)(b * Hn + h) * Sn + st * 16 + m) * DKn + nt * 16 + l15] =
          (_Float16)(acc[nt][r] + bias);
    }
  }
}

// ---------- K/V projection (V stored transposed) ----------
__global__ __launch_bounds__(128) void kvproj_kernel(const float* __restrict__ key,
                                                     const float* __restrict__ value,
                                                     const _Float16* __restrict__ WkT,
                                                     const _Float16* __restrict__ WvT,
                                                     const float* __restrict__ bk,
                                                     const float* __restrict__ bv,
                                                     _Float16* __restrict__ kbuf,
                                                     _Float16* __restrict__ vT) {
  int lane = threadIdx.x & 31;
  int w    = blockIdx.x * 4 + (threadIdx.x >> 5);   // 0..255
  int st   = w & 63;
  int b    = w >> 6;
  int l15  = lane & 15;
  int hi8  = (lane & 16) ? 8 : 0;
  int kb   = (lane & 16) ? 16 : 0;

  // ---- K ----
  {
    const float* Arow = key + ((size_t)b * Sn + st * 16 + l15) * Dn;
    v8f acc[4];
#pragma unroll
    for (int nt = 0; nt < 4; ++nt) acc[nt] = zero8();
    for (int k0 = 0; k0 < Dn; k0 += 32) {
      v16h a = load_a_f32(Arow, k0, hi8);
#pragma unroll
      for (int nt = 0; nt < 4; ++nt)
        acc[nt] = wmma_f32_f16(a, load_b_f16(WkT + (size_t)(nt * 16 + l15) * Dn, k0, kb), acc[nt]);
    }
#pragma unroll
    for (int nt = 0; nt < 4; ++nt) {
      float bias = bk[nt * 16 + l15];
#pragma unroll
      for (int r = 0; r < 8; ++r) {
        int m = r + hi8;
        kbuf[((size_t)b * Sn + st * 16 + m) * DKn + nt * 16 + l15] =
            (_Float16)(acc[nt][r] + bias);
      }
    }
  }
  // ---- V (store transposed [B][DK][S]) ----
  {
    const float* Arow = value + ((size_t)b * Sn + st * 16 + l15) * Dn;
    v8f acc[4];
#pragma unroll
    for (int nt = 0; nt < 4; ++nt) acc[nt] = zero8();
    for (int k0 = 0; k0 < Dn; k0 += 32) {
      v16h a = load_a_f32(Arow, k0, hi8);
#pragma unroll
      for (int nt = 0; nt < 4; ++nt)
        acc[nt] = wmma_f32_f16(a, load_b_f16(WvT + (size_t)(nt * 16 + l15) * Dn, k0, kb), acc[nt]);
    }
#pragma unroll
    for (int nt = 0; nt < 4; ++nt) {
      float bias = bv[nt * 16 + l15];
#pragma unroll
      for (int r = 0; r < 8; ++r) {
        int m = r + hi8;
        vT[((size_t)b * DKn + nt * 16 + l15) * Sn + st * 16 + m] =
            (_Float16)(acc[nt][r] + bias);
      }
    }
  }
}

// ---------- attention: scores -> softmax (writes attn fp32) -> P@V ----------
// 4 waves/block; per wave: 32KB f16 score/prob buffer + rsum array in dynamic LDS.
constexpr int    ATTN_WAVES    = 4;
constexpr size_t LDS_PER_WAVE  = 16 * 1024 * 2 /*scores f16*/ + 256 /*rsum pad*/;
constexpr size_t ATTN_LDS      = ATTN_WAVES * LDS_PER_WAVE;   // 132,096 B (<320KB WGP)

__global__ __launch_bounds__(128) void attn_kernel(const _Float16* __restrict__ qh,
                                                   const _Float16* __restrict__ kbuf,
                                                   const _Float16* __restrict__ vT,
                                                   _Float16* __restrict__ concat,
                                                   float* __restrict__ attn_out) {
  extern __shared__ char smem[];
  int lane = threadIdx.x & 31;
  int wv   = threadIdx.x >> 5;
  _Float16* sc    = (_Float16*)(smem + (size_t)wv * LDS_PER_WAVE);
  float*    rsumA = (float*)(smem + (size_t)wv * LDS_PER_WAVE + 32768);

  int w   = blockIdx.x * ATTN_WAVES + wv;           // 0..4095
  int st  = w & 63;
  int h   = (w >> 6) & 15;
  int b   = w >> 10;
  int l15 = lane & 15;
  int hi8 = (lane & 16) ? 8 : 0;
  int kb  = (lane & 16) ? 16 : 0;

  // ---- phase 1: scores[16][1024] = (q_tile @ K^T) * scale, into LDS as f16 ----
  const _Float16* Aq = qh + ((size_t)(b * Hn + h) * Sn + st * 16 + l15) * DKn;
  v16h aq0 = load_a_f16(Aq, 0, hi8);
  v16h aq1 = load_a_f16(Aq, 32, hi8);

  for (int tt = 0; tt < 64; ++tt) {
    const _Float16* Bk = kbuf + ((size_t)b * Sn + tt * 16 + l15) * DKn;
    if (tt + 1 < 64)
      __builtin_prefetch(kbuf + ((size_t)b * Sn + (tt + 1) * 16 + l15) * DKn, 0, 0);
    v8f s = wmma_f32_f16(aq0, load_b_f16(Bk, 0, kb), zero8());
    s     = wmma_f32_f16(aq1, load_b_f16(Bk, 32, kb), s);
#pragma unroll
    for (int r = 0; r < 8; ++r)
      sc[(r + hi8) * Sn + tt * 16 + l15] = (_Float16)(s[r] * 0.125f);
  }
  __syncthreads();

  // ---- phase 2a: per-lane (row = lane&15, half = lane>>4) max, exp-in-place, sum ----
  {
    _Float16* myrow = sc + l15 * Sn + (lane >> 4) * 512;
    float mx = -1e30f;
    for (int j0 = 0; j0 < 512; j0 += 8) {
      v8h v = *(const v8h*)(myrow + j0);
#pragma unroll
      for (int i = 0; i < 8; ++i) mx = fmaxf(mx, (float)v[i]);
    }
    mx = fmaxf(mx, __shfl_xor(mx, 16, 32));
    float sum = 0.0f;
    for (int j0 = 0; j0 < 512; j0 += 8) {
      v8h v = *(const v8h*)(myrow + j0);
      v8h e;
#pragma unroll
      for (int i = 0; i < 8; ++i) {
        float ex = __expf((float)v[i] - mx);
        e[i] = (_Float16)ex;
        sum += ex;
      }
      *(v8h*)(myrow + j0) = e;
    }
    sum += __shfl_xor(sum, 16, 32);
    if (lane < 16) rsumA[l15] = 1.0f / sum;
  }
  __syncthreads();

  // ---- phase 2b: normalize in place (f16) + coalesced fp32 attn writes ----
  {
    float* abase = attn_out + ((size_t)(b * Hn + h) * Sn + st * 16) * Sn;
    for (int m = 0; m < 16; ++m) {
      float rs = rsumA[m];
      float*    adst = abase + (size_t)m * Sn;
      _Float16* srow = sc + m * Sn;
      for (int t0 = 0; t0 < Sn; t0 += 32) {
        float p = (float)srow[t0 + lane] * rs;
        adst[t0 + lane] = p;
        srow[t0 + lane] = (_Float16)p;
      }
    }
  }
  __syncthreads();

  // ---- phase 3: out_tile[16][64] = P[16][1024] @ V[1024][64] ----
  const _Float16* Ap = sc + l15 * Sn;
  const _Float16* Bv[4];
#pragma unroll
  for (int nt = 0; nt < 4; ++nt)
    Bv[nt] = vT + ((size_t)b * DKn + nt * 16 + l15) * Sn;

  v8f acc[4];
#pragma unroll
  for (int nt = 0; nt < 4; ++nt) acc[nt] = zero8();

  for (int k0 = 0; k0 < Sn; k0 += 32) {
    v16h a = load_a_f16(Ap, k0, hi8);   // ds_read_b128 x2
#pragma unroll
    for (int nt = 0; nt < 4; ++nt)
      acc[nt] = wmma_f32_f16(a, load_b_f16(Bv[nt], k0, kb), acc[nt]);
  }

#pragma unroll
  for (int nt = 0; nt < 4; ++nt) {
#pragma unroll
    for (int r = 0; r < 8; ++r) {
      int m = r + hi8;
      concat[((size_t)b * Sn + st * 16 + m) * (Hn * DKn) + h * DKn + nt * 16 + l15] =
          (_Float16)acc[nt][r];
    }
  }
}

// ---------- combine heads: out = concat[S,1024] @ Wc + bc (fp32 out) ----------
__global__ __launch_bounds__(128) void combine_kernel(const _Float16* __restrict__ concat,
                                                      const _Float16* __restrict__ WcT,
                                                      const float* __restrict__ bc,
                                                      float* __restrict__ out) {
  int lane = threadIdx.x & 31;
  int w    = blockIdx.x * 4 + (threadIdx.x >> 5);   // 0..255
  int st   = w & 63;
  int b    = w >> 6;
  int l15  = lane & 15;
  int hi8  = (lane & 16) ? 8 : 0;
  int kb   = (lane & 16) ? 16 : 0;

  const _Float16* Arow = concat + ((size_t)b * Sn + st * 16 + l15) * (Hn * DKn);
  const _Float16* Bcol[4];
#pragma unroll
  for (int nt = 0; nt < 4; ++nt)
    Bcol[nt] = WcT + (size_t)(nt * 16 + l15) * (Hn * DKn);

  v8f acc[4];
#pragma unroll
  for (int nt = 0; nt < 4; ++nt) acc[nt] = zero8();

  for (int k0 = 0; k0 < Hn * DKn; k0 += 32) {
    v16h a = load_a_f16(Arow, k0, hi8);
#pragma unroll
    for (int nt = 0; nt < 4; ++nt)
      acc[nt] = wmma_f32_f16(a, load_b_f16(Bcol[nt], k0, kb), acc[nt]);
  }

#pragma unroll
  for (int nt = 0; nt < 4; ++nt) {
    float bias = bc[nt * 16 + l15];
#pragma unroll
    for (int r = 0; r < 8; ++r) {
      int m = r + hi8;
      out[((size_t)b * Sn + st * 16 + m) * DKn + nt * 16 + l15] = acc[nt][r] + bias;
    }
  }
}

// ---------- host launch ----------
extern "C" void kernel_launch(void* const* d_in, const int* in_sizes, int n_in,
                              void* d_out, int out_size, void* d_ws, size_t ws_size,
                              hipStream_t stream) {
  const float* query = (const float*)d_in[0];
  const float* key   = (const float*)d_in[1];
  const float* value = (const float*)d_in[2];
  const float* Wq    = (const float*)d_in[3];
  const float* bq    = (const float*)d_in[4];
  const float* Wk    = (const float*)d_in[5];
  const float* bk    = (const float*)d_in[6];
  const float* Wv    = (const float*)d_in[7];
  const float* bv    = (const float*)d_in[8];
  const float* Wc    = (const float*)d_in[9];
  const float* bc    = (const float*)d_in[10];

  float* out      = (float*)d_out;
  float* attn_out = out + (size_t)Bn * Sn * DKn;

  char* ws = (char*)d_ws;
  _Float16* WqT    = (_Float16*)(ws + OFF_WQT);
  _Float16* WkT    = (_Float16*)(ws + OFF_WKT);
  _Float16* WvT    = (_Float16*)(ws + OFF_WVT);
  _Float16* WcT    = (_Float16*)(ws + OFF_WCT);
  _Float16* qh     = (_Float16*)(ws + OFF_QH);
  _Float16* kbuf   = (_Float16*)(ws + OFF_K);
  _Float16* vTb    = (_Float16*)(ws + OFF_VT);
  _Float16* concat = (_Float16*)(ws + OFF_CC);

  // weight transpose+convert (all are [*,1024,64] -> [*,64,1024])
  {
    long long tq = (long long)Hn * Dn * DKn;
    transpose_to_f16_kernel<<<(int)((tq + 255) / 256), 256, 0, stream>>>(Wq, WqT, Dn, DKn, tq);
    long long ts = (long long)Dn * DKn;
    transpose_to_f16_kernel<<<(int)((ts + 255) / 256), 256, 0, stream>>>(Wk, WkT, Dn, DKn, ts);
    transpose_to_f16_kernel<<<(int)((ts + 255) / 256), 256, 0, stream>>>(Wv, WvT, Dn, DKn, ts);
    transpose_to_f16_kernel<<<(int)((ts + 255) / 256), 256, 0, stream>>>(Wc, WcT, Hn * DKn, DKn, ts);
  }

  // projections
  qproj_kernel <<<1024, 128, 0, stream>>>(query, WqT, bq, qh);
  kvproj_kernel<<<  64, 128, 0, stream>>>(key, value, WkT, WvT, bk, bv, kbuf, vTb);

  // attention (dynamic LDS: 4 waves x 32KB score buffers)
  attn_kernel<<<1024, 128, ATTN_LDS, stream>>>(qh, kbuf, vTb, concat, attn_out);

  // combine heads
  combine_kernel<<<64, 128, 0, stream>>>(concat, WcT, bc, out);
}
